// FastSelfAttention_20718922236079
// MI455X (gfx1250) — matrix operand order repaired and checked by reading, not compiled, not run on hardware
//
#include <hip/hip_runtime.h>
#include <hip/hip_bf16.h>

typedef __attribute__((ext_vector_type(16))) __bf16 v16bf;
typedef __attribute__((ext_vector_type(8)))  float  v8f;
typedef int v4i __attribute__((vector_size(16)));

#define TILE_M 128
#define TILE_N 128
#define TILE_K 32

#if defined(__HIP_DEVICE_COMPILE__) && \
    __has_builtin(__builtin_amdgcn_global_load_async_to_lds_b128) && \
    __has_builtin(__builtin_amdgcn_s_wait_asynccnt)
#define ASYNC_LDS 1
#else
#define ASYNC_LDS 0
#endif

static __device__ __forceinline__ unsigned short f2bf(float f) {
    unsigned int u = __float_as_uint(f);
    unsigned int r = u + 0x7FFFu + ((u >> 16) & 1u);   // round-to-nearest-even
    return (unsigned short)(r >> 16);
}

union FragBF { uint4 u[2]; v16bf v; };

#if ASYNC_LDS
typedef __attribute__((address_space(1))) v4i* gptr_v4i;
typedef __attribute__((address_space(3))) v4i* lptr_v4i;
// 16-byte global -> LDS async DMA (ASYNCcnt-tracked), gfx1250 path.
static __device__ __forceinline__ void async_cp16(const unsigned short* g,
                                                  unsigned short* l) {
    __builtin_amdgcn_global_load_async_to_lds_b128(
        (gptr_v4i)(unsigned long long)g,
        (lptr_v4i)(unsigned long long)l,
        0, 0);
}
#endif

// ---------------------------------------------------------------------------
// LayerNorm over D, emit bf16 activations for the WMMA GEMMs.
// One block (256 thr = 8 waves) per row of [B*S, D].
// ---------------------------------------------------------------------------
__global__ __launch_bounds__(256)
void ln_bf16_kernel(const float* __restrict__ x, const float* __restrict__ g,
                    const float* __restrict__ bta, unsigned short* __restrict__ hbf,
                    int D) {
    const int row = blockIdx.x;
    const size_t base = (size_t)row * D;
    const int tid = threadIdx.x;

    float vals[8];
    float s = 0.f, s2 = 0.f;
#pragma unroll
    for (int i = 0; i < 8; ++i) {
        float v = x[base + tid + i * 256];
        vals[i] = v;
        s += v;
        s2 += v * v;
    }
#pragma unroll
    for (int o = 16; o > 0; o >>= 1) {
        s  += __shfl_xor(s,  o);
        s2 += __shfl_xor(s2, o);
    }
    __shared__ float redS[8], redS2[8];
    __shared__ float s_mu, s_rstd;
    const int wave = tid >> 5;
    if ((tid & 31) == 0) { redS[wave] = s; redS2[wave] = s2; }
    __syncthreads();
    if (tid == 0) {
        float S = 0.f, S2 = 0.f;
#pragma unroll
        for (int w = 0; w < 8; ++w) { S += redS[w]; S2 += redS2[w]; }
        float mu  = S / (float)D;
        float var = S2 / (float)D - mu * mu;
        s_mu = mu;
        s_rstd = rsqrtf(var + 1e-5f);
    }
    __syncthreads();
    const float mu = s_mu, rstd = s_rstd;
#pragma unroll
    for (int i = 0; i < 8; ++i) {
        int d = tid + i * 256;
        float h = (vals[i] - mu) * rstd * g[d] + bta[d];
        hbf[base + d] = f2bf(h);
    }
}

// ---------------------------------------------------------------------------
// fp32 -> bf16 weight conversion (grid-stride)
// ---------------------------------------------------------------------------
__global__ __launch_bounds__(256)
void cvt_bf16_kernel(const float* __restrict__ w, unsigned short* __restrict__ o, int n) {
    for (int i = blockIdx.x * 256 + threadIdx.x; i < n; i += gridDim.x * 256)
        o[i] = f2bf(w[i]);
}

// ---------------------------------------------------------------------------
// C[M,N] = A[M,K](bf16) @ W[N,K](bf16)^T + bias   via v_wmma_f32_16x16x32_bf16
// Block: 256 thr (8 waves, 4x2 over 128x128 tile). Each wave: 2x4 16x16 tiles.
// Double-buffered LDS; tile fill uses global_load_async_to_lds_b128 when the
// toolchain exposes the gfx1250 async builtins (ASYNCcnt-tracked DMA).
// ---------------------------------------------------------------------------
__global__ __launch_bounds__(256)
void gemm_bf16_wmma_kernel(const unsigned short* __restrict__ A,
                           const unsigned short* __restrict__ W,
                           const float* __restrict__ bias,
                           float* Cf, unsigned short* Cbf,
                           int M, int N, int K) {
    __shared__ __align__(16) unsigned short As[2][TILE_M * TILE_K];
    __shared__ __align__(16) unsigned short Ws[2][TILE_N * TILE_K];

    const int tid  = threadIdx.x;
    const int lane = tid & 31;
    const int wave = tid >> 5;
    const int mBlock = blockIdx.y * TILE_M;
    const int nBlock = blockIdx.x * TILE_N;
    const int wm = (wave & 3) * 32;   // wave's M offset in tile
    const int wn = (wave >> 2) * 64;  // wave's N offset in tile

    const int halfSel = lane >> 4;    // 0: lanes 0-15, 1: lanes 16-31
    const int lrow    = lane & 15;
    const int kbA = halfSel * 8;      // A frag K-base per ISA 16-bit A layout
    const int kbB = halfSel * 8;      // B frag K-base (same K order as A)

    // ---- per-thread staging coordinates (two 8-elem chunks per thread) ----
    const int r0 = tid >> 2,        k0 = (tid & 3) * 8;
    const int r1 = (tid + 256) >> 2, k1 = ((tid + 256) & 3) * 8;
    const unsigned short* gA0 = A + (size_t)(mBlock + r0) * K + k0;
    const unsigned short* gA1 = A + (size_t)(mBlock + r1) * K + k1;
    const unsigned short* gW0 = W + (size_t)(nBlock + r0) * K + k0;
    const unsigned short* gW1 = W + (size_t)(nBlock + r1) * K + k1;
    unsigned short* lA0[2] = { &As[0][r0 * TILE_K + k0], &As[1][r0 * TILE_K + k0] };
    unsigned short* lA1[2] = { &As[0][r1 * TILE_K + k1], &As[1][r1 * TILE_K + k1] };
    unsigned short* lW0[2] = { &Ws[0][r0 * TILE_K + k0], &Ws[1][r0 * TILE_K + k0] };
    unsigned short* lW1[2] = { &Ws[0][r1 * TILE_K + k1], &Ws[1][r1 * TILE_K + k1] };

    auto stage = [&](int buf, int kt) {
#if ASYNC_LDS
        async_cp16(gA0 + kt, lA0[buf]);
        async_cp16(gA1 + kt, lA1[buf]);
        async_cp16(gW0 + kt, lW0[buf]);
        async_cp16(gW1 + kt, lW1[buf]);
#else
        *(uint4*)lA0[buf] = *(const uint4*)(gA0 + kt);
        *(uint4*)lA1[buf] = *(const uint4*)(gA1 + kt);
        *(uint4*)lW0[buf] = *(const uint4*)(gW0 + kt);
        *(uint4*)lW1[buf] = *(const uint4*)(gW1 + kt);
#endif
    };

    v8f acc[2][4];
#pragma unroll
    for (int mi = 0; mi < 2; ++mi)
#pragma unroll
        for (int ni = 0; ni < 4; ++ni)
            acc[mi][ni] = (v8f){0.f, 0.f, 0.f, 0.f, 0.f, 0.f, 0.f, 0.f};

    // prologue: fill buffer 0
    int buf = 0;
    stage(0, 0);
#if ASYNC_LDS
    __builtin_amdgcn_s_wait_asynccnt(0);
#endif
    __syncthreads();

    for (int kt = 0; kt < K; kt += TILE_K) {
        // kick off async fill of the other buffer while we compute
        if (kt + TILE_K < K) stage(buf ^ 1, kt + TILE_K);

        // ---- gather fragments (contiguous 16B LDS reads per ISA layout) ----
        v16bf af[2], bfr[4];
#pragma unroll
        for (int mi = 0; mi < 2; ++mi) {
            const unsigned short* p = &As[buf][(wm + mi * 16 + lrow) * TILE_K];
            FragBF f;
            f.u[0] = *(const uint4*)&p[kbA];
            f.u[1] = *(const uint4*)&p[16 + kbA];
            af[mi] = f.v;
        }
#pragma unroll
        for (int ni = 0; ni < 4; ++ni) {
            const unsigned short* p = &Ws[buf][(wn + ni * 16 + lrow) * TILE_K];
            FragBF f;
            f.u[0] = *(const uint4*)&p[kbB];
            f.u[1] = *(const uint4*)&p[16 + kbB];
            bfr[ni] = f.v;
        }
#pragma unroll
        for (int mi = 0; mi < 2; ++mi)
#pragma unroll
            for (int ni = 0; ni < 4; ++ni)
                acc[mi][ni] = __builtin_amdgcn_wmma_f32_16x16x32_bf16(
                    false, af[mi], false, bfr[ni], (short)0, acc[mi][ni],
                    false, false);

#if ASYNC_LDS
        __builtin_amdgcn_s_wait_asynccnt(0);   // next tile landed in LDS
#endif
        __syncthreads();
        buf ^= 1;
    }

    // ---- epilogue: bias add + store (C layout: VGPR v -> M=v / M=8+v) ----
#pragma unroll
    for (int mi = 0; mi < 2; ++mi) {
#pragma unroll
        for (int ni = 0; ni < 4; ++ni) {
            int ncol  = nBlock + wn + ni * 16 + lrow;
            int mrow0 = mBlock + wm + mi * 16 + halfSel * 8;
            float bb = bias[ncol];
#pragma unroll
            for (int v = 0; v < 8; ++v) {
                float val = acc[mi][ni][v] + bb;
                size_t idx = (size_t)(mrow0 + v) * N + ncol;
                if (Cf)  Cf[idx]  = val;
                if (Cbf) Cbf[idx] = f2bf(val);
            }
        }
    }
}

// ---------------------------------------------------------------------------
// Scan 1: qw = exp(qa*scale); qw /= cumsum(qw); pooled = cumsum(qw*query);
//         mixed = pooled * keys.  One thread per (b,d) channel, coalesced in d.
// mixed_f32 may alias keys (element-wise in-place, read-before-write).
// ---------------------------------------------------------------------------
__global__ __launch_bounds__(256)
void scan1_kernel(const float* __restrict__ qa, const float* __restrict__ query,
                  const float* keys, float* mixed_f32,
                  unsigned short* __restrict__ mixed_bf,
                  int S, int D, float scale) {
    const int col = blockIdx.x * 256 + threadIdx.x;   // over B*D
    const int b = col / D, d = col % D;
    size_t idx = (size_t)b * S * D + d;
    float sumE = 0.f, pooled = 0.f;
    for (int s = 0; s < S; ++s, idx += D) {
        float e = __expf(qa[idx] * scale);
        sumE += e;
        pooled += (e / sumE) * query[idx];
        float mk = pooled * keys[idx];
        mixed_f32[idx] = mk;
        mixed_bf[idx]  = f2bf(mk);
    }
}

// ---------------------------------------------------------------------------
// Scan 2: kw chain over mixed_keys, then out = pooled_keys * values
// ---------------------------------------------------------------------------
__global__ __launch_bounds__(256)
void scan2_kernel(const float* __restrict__ ka, const float* __restrict__ mixed,
                  const float* __restrict__ values, float* __restrict__ out,
                  int S, int D, float scale) {
    const int col = blockIdx.x * 256 + threadIdx.x;
    const int b = col / D, d = col % D;
    size_t idx = (size_t)b * S * D + d;
    float sumE = 0.f, pooled = 0.f;
    for (int s = 0; s < S; ++s, idx += D) {
        float e = __expf(ka[idx] * scale);
        sumE += e;
        pooled += (e / sumE) * mixed[idx];
        out[idx] = pooled * values[idx];
    }
}

// ---------------------------------------------------------------------------
extern "C" void kernel_launch(void* const* d_in, const int* in_sizes, int n_in,
                              void* d_out, int out_size, void* d_ws, size_t ws_size,
                              hipStream_t stream) {
    (void)in_sizes; (void)n_in; (void)out_size; (void)ws_size;
    const int Bv = 2, S = 4096, D = 2048;
    const int M = Bv * S;            // 8192 rows
    const float scale = 0.022097086912079608f;  // 1/sqrt(2048)

    const float* hidden = (const float*)d_in[0];
    // d_in[1] = attention_mask (unused by forward)
    const float* ln_g = (const float*)d_in[2];
    const float* ln_b = (const float*)d_in[3];
    const float* Wq  = (const float*)d_in[4];  const float* bq  = (const float*)d_in[5];
    const float* Wqa = (const float*)d_in[6];  const float* bqa = (const float*)d_in[7];
    const float* Wk  = (const float*)d_in[8];  const float* bk  = (const float*)d_in[9];
    const float* Wka = (const float*)d_in[10]; const float* bka = (const float*)d_in[11];
    const float* Wv  = (const float*)d_in[12]; const float* bv  = (const float*)d_in[13];
    float* out = (float*)d_out;

    // ---- workspace layout (bytes) ----
    char* ws = (char*)d_ws;
    const size_t szActBf = (size_t)M * D * 2;      // 32 MB
    const size_t szWBf   = (size_t)D * D * 2;      //  8 MB
    const size_t szF32   = (size_t)M * D * 4;      // 64 MB
    unsigned short* h_bf   = (unsigned short*)(ws);                      // later reused as mixed_bf
    unsigned short* q_bf   = (unsigned short*)(ws + szActBf);
    unsigned short* Wq_bf  = (unsigned short*)(ws + 2 * szActBf);
    unsigned short* Wk_bf  = (unsigned short*)(ws + 2 * szActBf + 1 * szWBf);
    unsigned short* Wv_bf  = (unsigned short*)(ws + 2 * szActBf + 2 * szWBf);
    unsigned short* Wqa_bf = (unsigned short*)(ws + 2 * szActBf + 3 * szWBf);
    unsigned short* Wka_bf = (unsigned short*)(ws + 2 * szActBf + 4 * szWBf);
    char* fbase = ws + 2 * szActBf + 5 * szWBf;
    float* query_f  = (float*)(fbase);
    float* keys_f   = (float*)(fbase + 1 * szF32);   // overwritten in-place with mixed_keys
    float* values_f = (float*)(fbase + 2 * szF32);
    float* qa_f     = (float*)(fbase + 3 * szF32);   // reused for ka

    // 1) LayerNorm -> bf16 activations
    ln_bf16_kernel<<<M, 256, 0, stream>>>(hidden, ln_g, ln_b, h_bf, D);

    // 2) weight conversions fp32 -> bf16
    const int wElems = D * D;
    cvt_bf16_kernel<<<2048, 256, 0, stream>>>(Wq,  Wq_bf,  wElems);
    cvt_bf16_kernel<<<2048, 256, 0, stream>>>(Wk,  Wk_bf,  wElems);
    cvt_bf16_kernel<<<2048, 256, 0, stream>>>(Wv,  Wv_bf,  wElems);
    cvt_bf16_kernel<<<2048, 256, 0, stream>>>(Wqa, Wqa_bf, wElems);
    cvt_bf16_kernel<<<2048, 256, 0, stream>>>(Wka, Wka_bf, wElems);

    // 3) q/k/v projections (WMMA bf16). query also emitted in bf16 for the qa GEMM.
    dim3 ggrid(D / TILE_N, M / TILE_M);
    gemm_bf16_wmma_kernel<<<ggrid, 256, 0, stream>>>(h_bf, Wq_bf, bq, query_f, q_bf, M, D, D);
    gemm_bf16_wmma_kernel<<<ggrid, 256, 0, stream>>>(h_bf, Wk_bf, bk, keys_f,  (unsigned short*)nullptr, M, D, D);
    gemm_bf16_wmma_kernel<<<ggrid, 256, 0, stream>>>(h_bf, Wv_bf, bv, values_f,(unsigned short*)nullptr, M, D, D);

    // 4) qa = query @ Wqa^T + bqa
    gemm_bf16_wmma_kernel<<<ggrid, 256, 0, stream>>>(q_bf, Wqa_bf, bqa, qa_f, (unsigned short*)nullptr, M, D, D);

    // 5) query pooling scan -> mixed_keys (f32 in-place over keys, bf16 reusing h_bf)
    scan1_kernel<<<(Bv * D) / 256, 256, 0, stream>>>(qa_f, query_f, keys_f, keys_f, h_bf, S, D, scale);

    // 6) ka = mixed_keys @ Wka^T + bka  (reuse qa buffer for ka)
    gemm_bf16_wmma_kernel<<<ggrid, 256, 0, stream>>>(h_bf, Wka_bf, bka, qa_f, (unsigned short*)nullptr, M, D, D);

    // 7) key pooling scan -> out = pooled_keys * values
    scan2_kernel<<<(Bv * D) / 256, 256, 0, stream>>>(qa_f, keys_f, values_f, out, S, D, scale);
}
